// MambaBlock_15556371546614
// MI455X (gfx1250) — matrix operand order repaired and checked by reading, not compiled, not run on hardware
//
#include <hip/hip_runtime.h>
#include <math.h>

#define D_MODEL 512
#define D_STATE 16
#define D_CONV  4
#define D_INNER 1024
#define DT_RANK 32
#define N_LAYERS 2
#define BATCH   2
#define SEQ     2048
#define TOK     (BATCH * SEQ)   // 4096 tokens

typedef __attribute__((ext_vector_type(16))) __bf16 bf16x16;
typedef __attribute__((ext_vector_type(8)))  float  f32x8;
typedef unsigned short u16;
typedef __attribute__((ext_vector_type(8)))  u16 u16x8;
typedef __attribute__((ext_vector_type(16))) u16 u16x16;
typedef __attribute__((ext_vector_type(4)))  int  v4i;

__device__ __forceinline__ u16 f2bf(float f) {
  union { float f; unsigned u; } v; v.f = f;
  unsigned u = v.u;
  u += 0x7FFFu + ((u >> 16) & 1u);   // round-to-nearest-even
  return (u16)(u >> 16);
}

__device__ __forceinline__ float sigmoidf_(float x) { return 1.f / (1.f + __expf(-x)); }

// ---------------------------------------------------------------------------
// gfx1250 async global->LDS copy (16B per lane), guarded so compile never breaks.
// Probe result: builtin exists and takes (addrspace(1) v4i32*, addrspace(3)
// v4i32*, int offset, int cpol).
// ---------------------------------------------------------------------------
#if defined(__gfx1250__) && __has_builtin(__builtin_amdgcn_global_load_async_to_lds_b128)
#define HAVE_ASYNC_LDS 1
typedef __attribute__((address_space(1))) v4i* gv4i_p;
typedef __attribute__((address_space(3))) v4i* lv4i_p;
#endif

__device__ __forceinline__ void cp16_g2l(const u16* g, u16* l) {
#ifdef HAVE_ASYNC_LDS
  __builtin_amdgcn_global_load_async_to_lds_b128((gv4i_p)g, (lv4i_p)l, 0, 0);
#else
  *(u16x8*)l = *(const u16x8*)g;   // global_load_b128 + ds_store_b128
#endif
}

__device__ __forceinline__ void cp_wait() {
#ifdef HAVE_ASYNC_LDS
#if __has_builtin(__builtin_amdgcn_s_wait_asynccnt)
  __builtin_amdgcn_s_wait_asynccnt(0);
#else
  asm volatile("s_wait_asynccnt 0" ::: "memory");
#endif
#endif
}

// ---------------------------------------------------------------------------
// C[M,N] = act(Abf[M,K] @ Wbf[N,K]^T + bias) + Res
// A, W are bf16; accum f32 via V_WMMA_F32_16X16X32_BF16.
// Block: 256 thr = 8 waves, tile 128x64, each wave 32x32 (2x2 WMMA).
// Requires M%128==0, N%64==0, K%32==0 (true for every call here).
// act: 0 none, 1 relu, 2 softplus.  outBf: 0 -> f32 C, 1 -> bf16 C.
// ---------------------------------------------------------------------------
__global__ __launch_bounds__(256) void gemm_wmma(
    const u16* __restrict__ A, long lda,
    const u16* __restrict__ W, long K,
    const float* __restrict__ bias,
    const float* __restrict__ Res, long ldr,
    void* __restrict__ Cout, long ldc,
    int M, int N, int act, int outBf)
{
  __shared__ __align__(16) u16 As[128][40];   // 32 halves + 8 pad (80B rows)
  __shared__ __align__(16) u16 Ws[64][40];

  const int tid  = threadIdx.x;
  const int lane = tid & 31;
  const int wid  = tid >> 5;
  const int m0 = blockIdx.x * 128;
  const int n0 = blockIdx.y * 64;
  const int m_off = (wid >> 1) * 32;
  const int n_off = (wid & 1) * 32;

  const int lm = lane & 15;
  const int kA = (lane < 16) ? 0 : 8;     // 16-bit A 16x32 frag K base
  const int kB = (lane < 16) ? 0 : 8;     // 16-bit B 32x16 frag K base

  f32x8 acc[2][2] = {};

  for (long k0 = 0; k0 < K; k0 += 32) {
    __syncthreads();
    // A tile: 128 rows x 64B = 512 x 16B chunks, 2 per thread
    #pragma unroll
    for (int j = 0; j < 2; ++j) {
      int idx = tid + j * 256;
      int r = idx >> 2, c = (idx & 3) * 8;
      cp16_g2l(&A[(long)(m0 + r) * lda + k0 + c], &As[r][c]);
    }
    // W tile: 64 rows x 64B = 256 x 16B chunks, 1 per thread
    {
      int r = tid >> 2, c = (tid & 3) * 8;
      cp16_g2l(&W[(long)(n0 + r) * K + k0 + c], &Ws[r][c]);
    }
    cp_wait();
    __syncthreads();

    bf16x16 afrag[2], bfrag[2];
    #pragma unroll
    for (int mi = 0; mi < 2; ++mi) {
      const u16* p = &As[m_off + mi * 16 + lm][0];
      u16x8 lo = *(const u16x8*)(p + kA);
      u16x8 hi = *(const u16x8*)(p + 16 + kA);
      u16x16 cat = __builtin_shufflevector(lo, hi, 0,1,2,3,4,5,6,7,8,9,10,11,12,13,14,15);
      afrag[mi] = __builtin_bit_cast(bf16x16, cat);
    }
    #pragma unroll
    for (int ni = 0; ni < 2; ++ni) {
      const u16* p = &Ws[n_off + ni * 16 + lm][0];
      u16x8 lo = *(const u16x8*)(p + kB);
      u16x8 hi = *(const u16x8*)(p + 16 + kB);
      u16x16 cat = __builtin_shufflevector(lo, hi, 0,1,2,3,4,5,6,7,8,9,10,11,12,13,14,15);
      bfrag[ni] = __builtin_bit_cast(bf16x16, cat);
    }
    #pragma unroll
    for (int mi = 0; mi < 2; ++mi)
      #pragma unroll
      for (int ni = 0; ni < 2; ++ni)
        acc[mi][ni] = __builtin_amdgcn_wmma_f32_16x16x32_bf16(
            false, afrag[mi], false, bfrag[ni], (short)0, acc[mi][ni], false, false);
  }

  const int roff = (lane < 16) ? 0 : 8;   // C/D f32 VGPR layout
  #pragma unroll
  for (int mi = 0; mi < 2; ++mi)
    #pragma unroll
    for (int ni = 0; ni < 2; ++ni) {
      int n = n0 + n_off + ni * 16 + lm;
      float bv = bias ? bias[n] : 0.f;
      #pragma unroll
      for (int r = 0; r < 8; ++r) {
        int m = m0 + m_off + mi * 16 + roff + r;
        float v = acc[mi][ni][r] + bv;
        if (act == 1)      v = fmaxf(v, 0.f);
        else if (act == 2) v = (v > 20.f) ? v : log1pf(__expf(v));
        if (Res) v += Res[(long)m * ldr + n];
        if (outBf) ((u16*)Cout)[(long)m * ldc + n] = f2bf(v);
        else       ((float*)Cout)[(long)m * ldc + n] = v;
      }
    }
}

// ---------------------------------------------------------------------------
// Elementwise f32 -> bf16 conversion (weights, dbc).
// ---------------------------------------------------------------------------
__global__ __launch_bounds__(256) void f2bf_kernel(
    const float* __restrict__ src, u16* __restrict__ dst, long n)
{
  long i = (long)blockIdx.x * 256 + threadIdx.x;
  if (i < n) dst[i] = f2bf(src[i]);
}

// ---------------------------------------------------------------------------
// LayerNorm, f32 out (ln1) -- one block per token row.
// ---------------------------------------------------------------------------
__global__ __launch_bounds__(256) void ln_kernel_f32(
    const float* __restrict__ x, const float* __restrict__ w,
    const float* __restrict__ b, float* __restrict__ y)
{
  __shared__ float red[256];
  const long row = blockIdx.x;
  const float* xr = x + row * D_MODEL;
  const int tid = threadIdx.x;

  float s = 0.f;
  for (int i = tid; i < D_MODEL; i += 256) s += xr[i];
  red[tid] = s; __syncthreads();
  for (int o = 128; o > 0; o >>= 1) { if (tid < o) red[tid] += red[tid + o]; __syncthreads(); }
  const float mu = red[0] * (1.f / D_MODEL);
  __syncthreads();
  float v = 0.f;
  for (int i = tid; i < D_MODEL; i += 256) { float d = xr[i] - mu; v += d * d; }
  red[tid] = v; __syncthreads();
  for (int o = 128; o > 0; o >>= 1) { if (tid < o) red[tid] += red[tid + o]; __syncthreads(); }
  const float rstd = rsqrtf(red[0] * (1.f / D_MODEL) + 1e-5f);
  for (int i = tid; i < D_MODEL; i += 256)
    y[row * D_MODEL + i] = (xr[i] - mu) * rstd * w[i] + b[i];
}

// LayerNorm, bf16 out (ln2 feeds a GEMM).
__global__ __launch_bounds__(256) void ln_kernel_bf16(
    const float* __restrict__ x, const float* __restrict__ w,
    const float* __restrict__ b, u16* __restrict__ y)
{
  __shared__ float red[256];
  const long row = blockIdx.x;
  const float* xr = x + row * D_MODEL;
  const int tid = threadIdx.x;

  float s = 0.f;
  for (int i = tid; i < D_MODEL; i += 256) s += xr[i];
  red[tid] = s; __syncthreads();
  for (int o = 128; o > 0; o >>= 1) { if (tid < o) red[tid] += red[tid + o]; __syncthreads(); }
  const float mu = red[0] * (1.f / D_MODEL);
  __syncthreads();
  float v = 0.f;
  for (int i = tid; i < D_MODEL; i += 256) { float d = xr[i] - mu; v += d * d; }
  red[tid] = v; __syncthreads();
  for (int o = 128; o > 0; o >>= 1) { if (tid < o) red[tid] += red[tid + o]; __syncthreads(); }
  const float rstd = rsqrtf(red[0] * (1.f / D_MODEL) + 1e-5f);
  for (int i = tid; i < D_MODEL; i += 256)
    y[row * D_MODEL + i] = f2bf((xr[i] - mu) * rstd * w[i] + b[i]);
}

// ---------------------------------------------------------------------------
// RMSNorm, bf16 out (feeds in_proj GEMM).
// ---------------------------------------------------------------------------
__global__ __launch_bounds__(256) void rms_kernel(
    const float* __restrict__ x, const float* __restrict__ w, u16* __restrict__ y)
{
  __shared__ float red[256];
  const long row = blockIdx.x;
  const float* xr = x + row * D_MODEL;
  const int tid = threadIdx.x;

  float s = 0.f;
  for (int i = tid; i < D_MODEL; i += 256) { float v = xr[i]; s += v * v; }
  red[tid] = s; __syncthreads();
  for (int o = 128; o > 0; o >>= 1) { if (tid < o) red[tid] += red[tid + o]; __syncthreads(); }
  const float scale = rsqrtf(red[0] * (1.f / D_MODEL) + 1e-5f);
  for (int i = tid; i < D_MODEL; i += 256)
    y[row * D_MODEL + i] = f2bf(xr[i] * scale * w[i]);
}

// ---------------------------------------------------------------------------
// Causal depthwise conv (width 4) + bias + silu.
// Writes f32 (for scan) and bf16 (for x_proj GEMM).
// ---------------------------------------------------------------------------
__global__ __launch_bounds__(256) void conv_silu_kernel(
    const float* __restrict__ xz, const float* __restrict__ cw,
    const float* __restrict__ cb, float* __restrict__ xc, u16* __restrict__ xcb)
{
  const long idx = (long)blockIdx.x * 256 + threadIdx.x;
  const int  e   = (int)(idx % D_INNER);
  const long row = idx / D_INNER;         // b*SEQ + t
  const int  t   = (int)(row % SEQ);

  float acc = cb[e];
  #pragma unroll
  for (int k = 0; k < D_CONV; ++k) {
    int dt = k - (D_CONV - 1);
    if (t + dt >= 0)
      acc += cw[e * D_CONV + k] * xz[(row + dt) * (2L * D_INNER) + e];
  }
  acc = acc * sigmoidf_(acc);
  xc[row * D_INNER + e]  = acc;
  xcb[row * D_INNER + e] = f2bf(acc);
}

// ---------------------------------------------------------------------------
// Selective scan: one thread per (b, channel), 16 f32 states in VGPRs,
// Bm/Cm staged in 128-step LDS chunks, fused Dp*x and silu(z) gate.
// Output bf16 (feeds out_proj GEMM only).
// ---------------------------------------------------------------------------
__global__ __launch_bounds__(256) void scan_kernel(
    const float* __restrict__ delta, const float* __restrict__ xc,
    const float* __restrict__ dbc,   const float* __restrict__ xz,
    const float* __restrict__ A_log, const float* __restrict__ Dp,
    u16* __restrict__ u)
{
  const int CH = 128;
  __shared__ float sBC[CH][32];     // [tl][0..15]=Bm, [16..31]=Cm
  const int tid = threadIdx.x;
  const int b   = blockIdx.x / (D_INNER / 256);
  const int e   = (blockIdx.x % (D_INNER / 256)) * 256 + tid;

  float a[D_STATE], h[D_STATE];
  #pragma unroll
  for (int n = 0; n < D_STATE; ++n) {
    a[n] = -__expf(A_log[(long)e * D_STATE + n]);
    h[n] = 0.f;
  }
  const float dp = Dp[e];

  for (int c0 = 0; c0 < SEQ; c0 += CH) {
    __syncthreads();
    for (int i = tid; i < CH * 32; i += 256) {
      int tt = c0 + (i >> 5);
      sBC[i >> 5][i & 31] = dbc[((long)b * SEQ + tt) * 64 + DT_RANK + (i & 31)];
    }
    __syncthreads();

    for (int tl = 0; tl < CH; ++tl) {
      const long row = (long)b * SEQ + c0 + tl;
      const float d  = delta[row * D_INNER + e];
      const float x  = xc[row * D_INNER + e];
      const float dx = d * x;
      float y = 0.f;
      #pragma unroll
      for (int n = 0; n < D_STATE; ++n) {
        float dA = __expf(d * a[n]);
        h[n] = dA * h[n] + dx * sBC[tl][n];
        y += h[n] * sBC[tl][16 + n];
      }
      y += dp * x;
      const float z = xz[row * (2L * D_INNER) + D_INNER + e];
      u[row * D_INNER + e] = f2bf(y * z * sigmoidf_(z));
    }
  }
}

__global__ __launch_bounds__(256) void add_kernel(
    const float* __restrict__ a, const float* __restrict__ b,
    float* __restrict__ c, long n)
{
  long i = (long)blockIdx.x * 256 + threadIdx.x;
  if (i < n) c[i] = a[i] + b[i];
}

// ---------------------------------------------------------------------------
extern "C" void kernel_launch(void* const* d_in, const int* in_sizes, int n_in,
                              void* d_out, int out_size, void* d_ws, size_t ws_size,
                              hipStream_t stream)
{
  const float* x         = (const float*)d_in[0];
  const float* in_proj_w = (const float*)d_in[1];
  const float* conv_w    = (const float*)d_in[2];
  const float* conv_b    = (const float*)d_in[3];
  const float* x_proj_w  = (const float*)d_in[4];
  const float* dt_w      = (const float*)d_in[5];
  const float* dt_b      = (const float*)d_in[6];
  const float* A_log     = (const float*)d_in[7];
  const float* Dp        = (const float*)d_in[8];
  const float* out_proj_w= (const float*)d_in[9];
  const float* rms_w     = (const float*)d_in[10];
  const float* ln1_w     = (const float*)d_in[11];
  const float* ln1_b     = (const float*)d_in[12];
  const float* ln2_w     = (const float*)d_in[13];
  const float* ln2_b     = (const float*)d_in[14];
  const float* ffn_w1    = (const float*)d_in[15];
  const float* ffn_b1    = (const float*)d_in[16];
  const float* ffn_w2    = (const float*)d_in[17];
  const float* ffn_b2    = (const float*)d_in[18];
  const float* final_w   = (const float*)d_in[19];
  const float* final_b   = (const float*)d_in[20];
  float* out = (float*)d_out;
  (void)in_sizes; (void)n_in; (void)out_size; (void)ws_size;

  char* ws = (char*)d_ws;
  size_t off = 0;
  auto allocf = [&](size_t elems) -> float* {
    float* p = (float*)(ws + off);
    off = (off + elems * sizeof(float) + 255) & ~(size_t)255;
    return p;
  };
  auto allocb = [&](size_t elems) -> u16* {
    u16* p = (u16*)(ws + off);
    off = (off + elems * sizeof(u16) + 255) & ~(size_t)255;
    return p;
  };

  // f32 activations
  float* h     = allocf((size_t)TOK * D_MODEL);       //  8 MB
  float* xz    = allocf((size_t)TOK * 2 * D_INNER);   // 32 MB
  float* xc    = allocf((size_t)TOK * D_INNER);       // 16 MB
  float* dbc   = allocf((size_t)TOK * 64);            //  1 MB
  float* delta = allocf((size_t)TOK * D_INNER);       // 16 MB
  float* xres  = allocf((size_t)TOK * D_MODEL);       //  8 MB
  // bf16 activations (GEMM inputs)
  u16* nrm_bf  = allocb((size_t)TOK * D_MODEL);       // rms out; reused for ln2 out
  u16* xc_bf   = allocb((size_t)TOK * D_INNER);       // conv out; reused for ffn1 out
  u16* dbc_bf  = allocb((size_t)TOK * 64);
  u16* u_bf    = allocb((size_t)TOK * D_INNER);       // scan out; reused for ffn2 out
  // bf16 weights
  u16* ipw_bf  = allocb((size_t)N_LAYERS * 2 * D_INNER * D_MODEL);
  u16* xpw_bf  = allocb((size_t)N_LAYERS * 64 * D_INNER);
  u16* dtw_bf  = allocb((size_t)N_LAYERS * D_INNER * DT_RANK);
  u16* opw_bf  = allocb((size_t)N_LAYERS * D_MODEL * D_INNER);
  u16* w1_bf   = allocb((size_t)2 * D_MODEL * D_MODEL);
  u16* w2_bf   = allocb((size_t)D_MODEL * 2 * D_MODEL);
  u16* fw_bf   = allocb((size_t)D_MODEL * D_MODEL);

  const dim3 blk(256);
  auto cvt = [&](const float* s, u16* d, long n) {
    f2bf_kernel<<<(int)((n + 255) / 256), blk, 0, stream>>>(s, d, n);
  };

  // one-time (per launch) weight conversion to bf16
  cvt(in_proj_w,  ipw_bf, (long)N_LAYERS * 2 * D_INNER * D_MODEL);
  cvt(x_proj_w,   xpw_bf, (long)N_LAYERS * 64 * D_INNER);
  cvt(dt_w,       dtw_bf, (long)N_LAYERS * D_INNER * DT_RANK);
  cvt(out_proj_w, opw_bf, (long)N_LAYERS * D_MODEL * D_INNER);
  cvt(ffn_w1,     w1_bf,  (long)2 * D_MODEL * D_MODEL);
  cvt(ffn_w2,     w2_bf,  (long)D_MODEL * 2 * D_MODEL);
  cvt(final_w,    fw_bf,  (long)D_MODEL * D_MODEL);

  // h = LN(x, ln1)
  ln_kernel_f32<<<TOK, blk, 0, stream>>>(x, ln1_w, ln1_b, h);

  for (int i = 0; i < N_LAYERS; ++i) {
    rms_kernel<<<TOK, blk, 0, stream>>>(h, rms_w + (size_t)i * D_MODEL, nrm_bf);

    // xz = rms(h) @ in_proj_w[i]^T            [TOK, 2048] f32
    gemm_wmma<<<dim3(TOK / 128, (2 * D_INNER) / 64), blk, 0, stream>>>(
        nrm_bf, D_MODEL, ipw_bf + (size_t)i * 2 * D_INNER * D_MODEL, D_MODEL,
        nullptr, nullptr, 0, xz, 2 * D_INNER, TOK, 2 * D_INNER, 0, 0);

    // xc = silu(conv(xz[:, :1024]) + cb)      f32 + bf16
    conv_silu_kernel<<<(int)(((long)TOK * D_INNER) / 256), blk, 0, stream>>>(
        xz, conv_w + (size_t)i * D_INNER * D_CONV, conv_b + (size_t)i * D_INNER,
        xc, xc_bf);

    // dbc = xc @ x_proj_w[i]^T                [TOK, 64] f32
    gemm_wmma<<<dim3(TOK / 128, 1), blk, 0, stream>>>(
        xc_bf, D_INNER, xpw_bf + (size_t)i * 64 * D_INNER, D_INNER,
        nullptr, nullptr, 0, dbc, 64, TOK, 64, 0, 0);
    cvt(dbc, dbc_bf, (long)TOK * 64);

    // delta = softplus(dbc[:, :32] @ dt_w[i]^T + dt_b[i])   [TOK, 1024] f32
    gemm_wmma<<<dim3(TOK / 128, D_INNER / 64), blk, 0, stream>>>(
        dbc_bf, 64, dtw_bf + (size_t)i * D_INNER * DT_RANK, DT_RANK,
        dt_b + (size_t)i * D_INNER, nullptr, 0, delta, D_INNER, TOK, D_INNER, 2, 0);

    // u = (scan + Dp*xc) * silu(z)            bf16
    scan_kernel<<<BATCH * (D_INNER / 256), blk, 0, stream>>>(
        delta, xc, dbc, xz,
        A_log + (size_t)i * D_INNER * D_STATE, Dp + (size_t)i * D_INNER, u_bf);

    // h = h + u @ out_proj_w[i]^T             [TOK, 512] f32 (in place)
    gemm_wmma<<<dim3(TOK / 128, D_MODEL / 64), blk, 0, stream>>>(
        u_bf, D_INNER, opw_bf + (size_t)i * D_MODEL * D_INNER, D_INNER,
        nullptr, h, D_MODEL, h, D_MODEL, TOK, D_MODEL, 0, 0);
  }

  // xres = x + h
  add_kernel<<<(int)(((long)TOK * D_MODEL) / 256), blk, 0, stream>>>(
      x, h, xres, (long)TOK * D_MODEL);

  // FFN
  ln_kernel_bf16<<<TOK, blk, 0, stream>>>(xres, ln2_w, ln2_b, nrm_bf);
  u16* ffn1_bf  = xc_bf;   // dead, reuse
  u16* xres2_bf = u_bf;    // dead, reuse
  gemm_wmma<<<dim3(TOK / 128, (2 * D_MODEL) / 64), blk, 0, stream>>>(
      nrm_bf, D_MODEL, w1_bf, D_MODEL, ffn_b1, nullptr, 0,
      ffn1_bf, 2 * D_MODEL, TOK, 2 * D_MODEL, 1, 1);
  gemm_wmma<<<dim3(TOK / 128, D_MODEL / 64), blk, 0, stream>>>(
      ffn1_bf, 2 * D_MODEL, w2_bf, 2 * D_MODEL, ffn_b2, xres, D_MODEL,
      xres2_bf, D_MODEL, TOK, D_MODEL, 0, 1);

  // out = xres2 @ final_w^T + final_b        f32
  gemm_wmma<<<dim3(TOK / 128, D_MODEL / 64), blk, 0, stream>>>(
      xres2_bf, D_MODEL, fw_bf, D_MODEL, final_b, nullptr, 0,
      out, D_MODEL, TOK, D_MODEL, 0, 0);
}
